// RWKV_RTM_66571993088388
// MI455X (gfx1250) — compile-verified, matrix operand order checked by reading, not run on hardware
//
#include <hip/hip_runtime.h>

// ---------------- problem constants ----------------
#define B_    4
#define HS_   48
#define C_    768
#define R_    192
#define TOPK_ 3
#define NH_   8
#define HD_   96          // C_/NH_
#define NW_   576
#define NS_   2304        // HS_*HS_
#define WTOK_ 4
#define TROWS_ 48         // B_*TOPK_*WTOK_
#define CROWS_ 9168       // B_*(NW_-TOPK_)*WTOK_
#define AROWS_ 9216       // B_*NS_

typedef __attribute__((ext_vector_type(16))) _Float16 v16h;
typedef __attribute__((ext_vector_type(8)))  float    v8f;
typedef __attribute__((ext_vector_type(4)))  int      v4i;

union FragU { v16h h; unsigned int u[8]; uint4 q[2]; };
union Pack8 { _Float16 h[8]; uint4 q; };

// async global->LDS staging (CDNA5), guarded so absence cannot break the build
#if __has_builtin(__builtin_amdgcn_global_load_async_to_lds_b128)
#define ASYNC_OK 1
typedef __attribute__((address_space(1))) v4i gv4i;   // global int4
typedef __attribute__((address_space(3))) v4i lv4i;   // LDS int4
#else
#define ASYNC_OK 0
#endif

__device__ __forceinline__ void wait_asynccnt0() {
#if __has_builtin(__builtin_amdgcn_s_wait_asynccnt)
  __builtin_amdgcn_s_wait_asynccnt(0);
#else
  asm volatile("s_wait_asynccnt 0x0" ::: "memory");
#endif
}

__device__ __forceinline__ int kb_of(int r, int half) {
  // CDNA5 16-bit A/B fragment K index for VGPR r, lane-half `half`
  return (r < 4) ? (half * 8 + 2 * r) : (16 + half * 8 + 2 * (r - 4));
}

// fragment from f16 row chunk: element 2r+p <-> k = kb_of(r,half)+p,
// i.e. two contiguous 8-halfword runs -> two b128 loads.
__device__ __forceinline__ v16h frag_from_f16(const _Float16* __restrict__ p, int half) {
  FragU f;
  f.q[0] = *(const uint4*)(p + half * 8);
  f.q[1] = *(const uint4*)(p + 16 + half * 8);
  return f.h;
}

// ============================================================
// f32 -> f16 bulk convert (8 elements / thread, b128 in, b128 out)
// ============================================================
__global__ __launch_bounds__(256) void cvt_f16_kernel(const float* __restrict__ src,
                                                      _Float16* __restrict__ dst, int n)
{
  const int i = (blockIdx.x * 256 + threadIdx.x) * 8;
  if (i >= n) return;
  const float4 a = *(const float4*)(src + i);
  const float4 b = *(const float4*)(src + i + 4);
  Pack8 p;
  p.h[0] = (_Float16)a.x; p.h[1] = (_Float16)a.y; p.h[2] = (_Float16)a.z; p.h[3] = (_Float16)a.w;
  p.h[4] = (_Float16)b.x; p.h[5] = (_Float16)b.y; p.h[6] = (_Float16)b.z; p.h[7] = (_Float16)b.w;
  *(uint4*)(dst + i) = p.q;
}

// ============================================================
// WMMA GEMM:  Y[M,N] = gatherA(X)[M,K] @ W[N,K]^T (+bias)
// f16 inputs, f32 accumulate, 16x64 tile per wave.
// ============================================================
template <bool MAPA, bool MAPY, bool RESID, bool OUT16>
__global__ __launch_bounds__(128) void gemm_xwT_kernel(
    const _Float16* __restrict__ X, int ldx, const int* __restrict__ rowmapA,
    const _Float16* __restrict__ W, int ldw,
    const float* __restrict__ bias,
    void* __restrict__ Yv, int ldy, const int* __restrict__ rowmapY,
    int M, int N, int Kd,
    const float* __restrict__ resid, const float* __restrict__ scale_vec)
{
  const int wave = threadIdx.x >> 5;
  const int lane = threadIdx.x & 31;
  const int half = lane >> 4;
  const int l16  = lane & 15;

  const int tiles_n = N >> 6;                // 64-wide tiles
  const int tile = blockIdx.x * 4 + wave;
  if (tile >= (M >> 4) * tiles_n) return;
  const int mt = tile / tiles_n;
  const int nt = tile % tiles_n;

  const int arow = mt * 16 + l16;
  const int xrow = MAPA ? rowmapA[arow] : arow;
  const _Float16* __restrict__ xp  = X + (size_t)xrow * ldx;
  const _Float16* __restrict__ wp0 = W + (size_t)(nt * 64 +  0 + l16) * ldw;
  const _Float16* __restrict__ wp1 = W + (size_t)(nt * 64 + 16 + l16) * ldw;
  const _Float16* __restrict__ wp2 = W + (size_t)(nt * 64 + 32 + l16) * ldw;
  const _Float16* __restrict__ wp3 = W + (size_t)(nt * 64 + 48 + l16) * ldw;

  v8f acc0 = {}, acc1 = {}, acc2 = {}, acc3 = {};
  for (int k0 = 0; k0 < Kd; k0 += 32) {
    const v16h a  = frag_from_f16(xp  + k0, half);
    const v16h b0 = frag_from_f16(wp0 + k0, half);
    const v16h b1 = frag_from_f16(wp1 + k0, half);
    const v16h b2 = frag_from_f16(wp2 + k0, half);
    const v16h b3 = frag_from_f16(wp3 + k0, half);
    acc0 = __builtin_amdgcn_wmma_f32_16x16x32_f16(false, a, false, b0, (short)0, acc0, false, false);
    acc1 = __builtin_amdgcn_wmma_f32_16x16x32_f16(false, a, false, b1, (short)0, acc1, false, false);
    acc2 = __builtin_amdgcn_wmma_f32_16x16x32_f16(false, a, false, b2, (short)0, acc2, false, false);
    acc3 = __builtin_amdgcn_wmma_f32_16x16x32_f16(false, a, false, b3, (short)0, acc3, false, false);
  }

#pragma unroll
  for (int s = 0; s < 4; ++s) {
    const v8f acc = (s == 0) ? acc0 : (s == 1) ? acc1 : (s == 2) ? acc2 : acc3;
    const int col = nt * 64 + s * 16 + l16;
    const float bv = bias ? bias[col] : 0.f;
#pragma unroll
    for (int r = 0; r < 8; ++r) {
      const int row  = mt * 16 + r + 8 * half;
      const int orow = MAPY ? rowmapY[row] : row;
      float v = acc[r] + bv;
      if (RESID) v = resid[(size_t)orow * ldy + col] + scale_vec[col] * v;
      if (OUT16) ((_Float16*)Yv)[(size_t)orow * ldy + col] = (_Float16)v;
      else       ((float*)Yv)[(size_t)orow * ldy + col] = v;
    }
  }
}

// ============================================================
// routing statistics
// ============================================================
__global__ __launch_bounds__(128) void colmean_kernel(const float* __restrict__ xz,
                                                      float* __restrict__ zm)
{
  const int b   = blockIdx.x;
  const int col = blockIdx.y * 128 + threadIdx.x;
  const float* p = xz + (size_t)b * NS_ * C_ + col;
  float s = 0.f;
  for (int r = 0; r < NS_; ++r) s += p[(size_t)r * C_];
  zm[b * C_ + col] = s * (1.0f / NS_);
}

__global__ __launch_bounds__(256) void l2n_kernel(float* __restrict__ z)
{
  __shared__ float red[256];
  const int b = blockIdx.x, tid = threadIdx.x;
  float s = 0.f;
  for (int c = tid; c < C_; c += 256) { float v = z[b * C_ + c]; s += v * v; }
  red[tid] = s; __syncthreads();
  for (int st = 128; st >= 1; st >>= 1) {
    if (tid < st) red[tid] += red[tid + st];
    __syncthreads();
  }
  const float inv = 1.0f / fmaxf(sqrtf(red[0]), 1e-12f);
  for (int c = tid; c < C_; c += 256) z[b * C_ + c] *= inv;
}

__global__ __launch_bounds__(256) void resp_kernel(const float* __restrict__ xs,
                                                   const float* __restrict__ zn,
                                                   float* __restrict__ resp)
{
  const int wid = blockIdx.x * 8 + (threadIdx.x >> 5);
  if (wid >= B_ * NS_) return;
  const int lane = threadIdx.x & 31;
  const int b = wid / NS_;
  const float* xp = xs + (size_t)wid * C_;
  const float* zp = zn + b * C_;
  float dot = 0.f, nn = 0.f;
  for (int c = lane; c < C_; c += 32) { float v = xp[c]; dot += v * zp[c]; nn += v * v; }
#pragma unroll
  for (int o = 16; o >= 1; o >>= 1) { dot += __shfl_xor(dot, o); nn += __shfl_xor(nn, o); }
  if (lane == 0) resp[wid] = dot / fmaxf(sqrtf(nn), 1e-12f);
}

__global__ __launch_bounds__(256) void route_kernel(const float* __restrict__ resp,
                                                    int* __restrict__ rm_t,
                                                    int* __restrict__ rm_c,
                                                    int* __restrict__ rmy_t,
                                                    int* __restrict__ rmy_c)
{
  __shared__ float wm[NW_];
  __shared__ int   ord[NW_];
  const int b = blockIdx.x, tid = threadIdx.x;
  const float* rp = resp + b * NS_;
  for (int w = tid; w < NW_; w += 256) {
    const int wi = w / 24, wj = w % 24;
    const float s = rp[(2 * wi) * HS_ + 2 * wj] + rp[(2 * wi) * HS_ + 2 * wj + 1] +
                    rp[(2 * wi + 1) * HS_ + 2 * wj] + rp[(2 * wi + 1) * HS_ + 2 * wj + 1];
    wm[w] = s * 0.25f;
  }
  __syncthreads();
  for (int i = tid; i < NW_; i += 256) {
    const float wiv = wm[i];
    int cnt = 0;
    for (int j = 0; j < NW_; ++j) {
      const float wjv = wm[j];
      cnt += (wjv > wiv) || (wjv == wiv && j < i);   // stable argsort(-w)
    }
    ord[cnt] = i;
  }
  __syncthreads();
  for (int r = tid; r < NS_; r += 256) {
    const int w = ord[r >> 2], t = r & 3;
    const int wi = w / 24, wj = w % 24, ti = t >> 1, tj = t & 1;
    const int src = b * NS_ + (wi * 2 + ti) * HS_ + (wj * 2 + tj);
    if (r < TOPK_ * WTOK_) {
      rm_t[b * 12 + r]  = src;
      rmy_t[b * 12 + r] = b * NS_ + r;
    } else {
      const int i = r - 12;
      rm_c[b * 2292 + i]  = src;
      rmy_c[b * 2292 + i] = b * NS_ + r;
    }
  }
}

// ============================================================
// RWKV single-step elementwise; y (-> next GEMM) f16, state f32.
// ============================================================
__global__ __launch_bounds__(256) void rwkv_ew_kernel(
    const float* __restrict__ kb, const float* __restrict__ vb,
    const float* __restrict__ rb, const float* __restrict__ st,
    const float* __restrict__ td, const float* __restrict__ tf,
    _Float16* __restrict__ y, float* __restrict__ ns, int rows)
{
  const int n = rows * R_;
  const int e = blockIdx.x * 256 + threadIdx.x;
  if (e >= n) return;
  const int j = e % R_;
  const float k = kb[e], v = vb[e];
  const float r = 1.f / (1.f + __expf(-rb[e]));
  const float aa = st[e], bb = st[n + e], pp = st[2 * n + e];
  const float w  = __expf(td[j]);
  const float ww = tf[j] + k;
  const float p  = fmaxf(pp, ww);
  const float e1 = __expf(pp - p), e2 = __expf(ww - p);
  const float wkv = (e1 * aa + e2 * v) / (e1 * bb + e2);
  const float ww2 = pp - w;
  const float p2  = fmaxf(ww2, k);
  const float f1 = __expf(ww2 - p2), f2 = __expf(k - p2);
  ns[e]         = f1 * aa + f2 * v;
  ns[n + e]     = f1 * bb + f2;
  ns[2 * n + e] = p2;
  y[e] = (_Float16)(r * wkv);
}

// LayerNorm (f32 in, f16 out -> feeds GEMMs only)
__global__ __launch_bounds__(256) void ln_kernel(const float* __restrict__ X,
                                                 const float* __restrict__ g,
                                                 const float* __restrict__ bta,
                                                 _Float16* __restrict__ Y, int rows)
{
  const int wid = blockIdx.x * 8 + (threadIdx.x >> 5);
  if (wid >= rows) return;
  const int lane = threadIdx.x & 31;
  const float* xp = X + (size_t)wid * C_;
  float s = 0.f, ss = 0.f;
  for (int c = lane; c < C_; c += 32) { float v = xp[c]; s += v; ss += v * v; }
#pragma unroll
  for (int o = 16; o >= 1; o >>= 1) { s += __shfl_xor(s, o); ss += __shfl_xor(ss, o); }
  const float m   = s * (1.0f / C_);
  const float var = ss * (1.0f / C_) - m * m;
  const float inv = rsqrtf(var + 1e-6f);
  _Float16* yp = Y + (size_t)wid * C_;
  for (int c = lane; c < C_; c += 32)
    yp[c] = (_Float16)((xp[c] - m) * inv * g[c] + bta[c]);
}

__global__ __launch_bounds__(256) void copy_tp_kernel(const float* __restrict__ tc,
                                                      float* __restrict__ tp)
{
  const int idx = blockIdx.x * 256 + threadIdx.x;
  if (idx >= B_ * 12 * C_) return;
  const int b = idx / (12 * C_), rem = idx % (12 * C_);
  tp[idx] = tc[(size_t)b * NS_ * C_ + rem];
}

// ============================================================
// Flash attention (all-f16). Block = 4 waves, one (b,h).
// K tile: async global->LDS b128 copies when available (ASYNCcnt),
// else manual b128 staging. V tile transposed via VGPR scatter.
// Fragments built with 32-bit LDS reads; next tile prefetched.
// ============================================================
#define KP_ 104
#define VP_ 36

__global__ __launch_bounds__(128) void attn_kernel(const _Float16* __restrict__ Q,
                                                   const _Float16* __restrict__ Kb,
                                                   const _Float16* __restrict__ Vb,
                                                   _Float16* __restrict__ O)
{
  __shared__ __align__(16) _Float16 Klds[32 * KP_];
  __shared__ __align__(16) _Float16 Vt[HD_ * VP_];
  __shared__ __align__(16) _Float16 pl[4][16][34];

  const int tid = threadIdx.x;
  const int wv = tid >> 5, lane = tid & 31;
  const int half = lane >> 4, l16 = lane & 15;
  const int qgrp = blockIdx.x % 36;
  const int bh   = blockIdx.x / 36;
  const int b = bh >> 3, h = bh & 7;
  const int qt = qgrp * 4 + wv;
  const size_t base = ((size_t)b * NS_) * C_ + h * HD_;

  const int srow = tid >> 2;          // staging: 4 threads per key row
  const int sseg = (tid & 3) * 24;    // 24 halfwords each

  v16h qf[3];
  {
    const _Float16* qrow = Q + base + (size_t)(qt * 16 + l16) * C_;
#pragma unroll
    for (int c3 = 0; c3 < 3; ++c3) qf[c3] = frag_from_f16(qrow + c3 * 32, half);
  }

  v8f acc[6]; float mrun[8], lrun[8];
#pragma unroll
  for (int j = 0; j < 6; ++j) acc[j] = (v8f){};
#pragma unroll
  for (int r = 0; r < 8; ++r) { mrun[r] = -1e30f; lrun[r] = 0.f; }
  const float sc = 0.1020620726159658f;   // 1/sqrt(96)

  const unsigned int* Ku = (const unsigned int*)Klds;
  const unsigned int* Vu = (const unsigned int*)Vt;
  const unsigned int* Pu = (const unsigned int*)&pl[wv][0][0];

  for (int kt = 0; kt < NS_ / 32; ++kt) {
    __syncthreads();                 // previous tile fully consumed
    {
      const _Float16* gk = Kb + base + (size_t)(kt * 32 + srow) * C_ + sseg;
#if ASYNC_OK
      gv4i* gsrc = (gv4i*)(unsigned long long)(size_t)gk;
      lv4i* ldst = (lv4i*)(unsigned)(size_t)&Klds[srow * KP_ + sseg];
      __builtin_amdgcn_global_load_async_to_lds_b128(gsrc, ldst, 0, 0);
      __builtin_amdgcn_global_load_async_to_lds_b128(gsrc, ldst, 16, 0);
      __builtin_amdgcn_global_load_async_to_lds_b128(gsrc, ldst, 32, 0);
#else
      *(uint4*)&Klds[srow * KP_ + sseg]      = ((const uint4*)gk)[0];
      *(uint4*)&Klds[srow * KP_ + sseg + 8]  = ((const uint4*)gk)[1];
      *(uint4*)&Klds[srow * KP_ + sseg + 16] = ((const uint4*)gk)[2];
#endif
      // stage V tile (transposed scatter)
      const _Float16* gv = Vb + base + (size_t)(kt * 32 + srow) * C_ + sseg;
#pragma unroll
      for (int i = 0; i < 3; ++i) {
        Pack8 p; p.q = ((const uint4*)gv)[i];
#pragma unroll
        for (int e = 0; e < 8; ++e)
          Vt[(sseg + i * 8 + e) * VP_ + srow] = p.h[e];
      }
      // prefetch next tile while this one is consumed
      const int ktn = (kt + 1 < NS_ / 32) ? kt + 1 : kt;
      __builtin_prefetch(Kb + base + (size_t)(ktn * 32 + srow) * C_ + sseg, 0, 1);
      __builtin_prefetch(Vb + base + (size_t)(ktn * 32 + srow) * C_ + sseg, 0, 1);
    }
#if ASYNC_OK
    wait_asynccnt0();
#endif
    __syncthreads();                 // tile staged

    // ---- S = Q @ K^T ----
    v8f s0 = {}, s1 = {};
#pragma unroll
    for (int c3 = 0; c3 < 3; ++c3) {
      FragU k0f, k1f;
#pragma unroll
      for (int r = 0; r < 8; ++r) {
        const int kw = (c3 * 32 + kb_of(r, half)) >> 1;
        k0f.u[r] = Ku[l16 * (KP_ / 2) + kw];
        k1f.u[r] = Ku[(16 + l16) * (KP_ / 2) + kw];
      }
      s0 = __builtin_amdgcn_wmma_f32_16x16x32_f16(false, qf[c3], false, k0f.h, (short)0, s0, false, false);
      s1 = __builtin_amdgcn_wmma_f32_16x16x32_f16(false, qf[c3], false, k1f.h, (short)0, s1, false, false);
    }

    // ---- online softmax: only the row-max is reduced per iteration ----
    float p0a[8], p1a[8], alpha[8];
#pragma unroll
    for (int r = 0; r < 8; ++r) {
      const float a = s0[r] * sc, b2 = s1[r] * sc;
      float mv = fmaxf(a, b2);
      mv = fmaxf(mv, __shfl_xor(mv, 1));
      mv = fmaxf(mv, __shfl_xor(mv, 2));
      mv = fmaxf(mv, __shfl_xor(mv, 4));
      mv = fmaxf(mv, __shfl_xor(mv, 8));
      const float mnew = fmaxf(mrun[r], mv);
      const float al = __expf(mrun[r] - mnew);
      const float p0 = __expf(a - mnew), p1 = __expf(b2 - mnew);
      lrun[r] = lrun[r] * al + (p0 + p1);   // per-lane partial sum
      mrun[r] = mnew;
      alpha[r] = al; p0a[r] = p0; p1a[r] = p1;
    }
#pragma unroll
    for (int j = 0; j < 6; ++j)
#pragma unroll
      for (int r = 0; r < 8; ++r) acc[j][r] *= alpha[r];

    // ---- transpose P (D layout -> A layout) through per-wave LDS ----
#pragma unroll
    for (int r = 0; r < 8; ++r) {
      pl[wv][r + 8 * half][l16]      = (_Float16)p0a[r];
      pl[wv][r + 8 * half][16 + l16] = (_Float16)p1a[r];
    }
    __builtin_amdgcn_wave_barrier();
    FragU ap;
#pragma unroll
    for (int r = 0; r < 8; ++r) ap.u[r] = Pu[l16 * 17 + (kb_of(r, half) >> 1)];
    __builtin_amdgcn_wave_barrier();

    // ---- O += P @ V ----
#pragma unroll
    for (int j = 0; j < 6; ++j) {
      FragU bv;
#pragma unroll
      for (int r = 0; r < 8; ++r)
        bv.u[r] = Vu[(j * 16 + l16) * (VP_ / 2) + (kb_of(r, half) >> 1)];
      acc[j] = __builtin_amdgcn_wmma_f32_16x16x32_f16(false, ap.h, false, bv.h, (short)0, acc[j], false, false);
    }
  }

  // epilogue: reduce deferred softmax denominators, write f16 O
#pragma unroll
  for (int r = 0; r < 8; ++r) {
    float l = lrun[r];
    l += __shfl_xor(l, 1); l += __shfl_xor(l, 2);
    l += __shfl_xor(l, 4); l += __shfl_xor(l, 8);
    const float inv = 1.f / l;
    _Float16* orow = O + base + (size_t)(qt * 16 + r + 8 * half) * C_;
#pragma unroll
    for (int j = 0; j < 6; ++j) orow[j * 16 + l16] = (_Float16)(acc[j][r] * inv);
  }
}

// ============================================================
// host-side orchestration
// ============================================================
static inline int gemm_blocks(int M, int N) { return ((M >> 4) * (N >> 6) + 3) / 4; }
static inline int cvt_blocks(int n) { return (n / 8 + 255) / 256; }

extern "C" void kernel_launch(void* const* d_in, const int* in_sizes, int n_in,
                              void* d_out, int out_size, void* d_ws, size_t ws_size,
                              hipStream_t stream) {
  (void)in_sizes; (void)n_in; (void)out_size; (void)ws_size;
  const float* xz       = (const float*)d_in[0];
  const float* xs       = (const float*)d_in[1];
  const float* state_t  = (const float*)d_in[2];
  const float* state_c  = (const float*)d_in[3];
  const float* Wdo_w    = (const float*)d_in[4];
  const float* Wdo_b    = (const float*)d_in[5];
  const float* Wdc_w    = (const float*)d_in[6];
  const float* Wdc_b    = (const float*)d_in[7];
  const float* Wup_w    = (const float*)d_in[8];
  const float* Wup_b    = (const float*)d_in[9];
  const float* te_td    = (const float*)d_in[10];
  const float* te_tf    = (const float*)d_in[11];
  const float* te_k     = (const float*)d_in[12];
  const float* te_v     = (const float*)d_in[13];
  const float* te_r     = (const float*)d_in[14];
  const float* te_o     = (const float*)d_in[15];
  const float* ce_td    = (const float*)d_in[16];
  const float* ce_tf    = (const float*)d_in[17];
  const float* ce_k     = (const float*)d_in[18];
  const float* ce_v     = (const float*)d_in[19];
  const float* ce_r     = (const float*)d_in[20];
  const float* ce_o     = (const float*)d_in[21];
  const float* qn_g     = (const float*)d_in[22];
  const float* qn_b     = (const float*)d_in[23];
  const float* fn_g     = (const float*)d_in[24];
  const float* fn_b     = (const float*)d_in[25];
  const float* attn_in_w  = (const float*)d_in[26];
  const float* attn_in_b  = (const float*)d_in[27];
  const float* attn_out_w = (const float*)d_in[28];
  const float* attn_out_b = (const float*)d_in[29];
  const float* router     = (const float*)d_in[30];

  // ---------- f32 workspace ----------
  float* ws = (float*)d_ws;
  size_t o = 0;
  float* zn   = ws + o; o += (size_t)B_ * C_;
  float* resp = ws + o; o += (size_t)B_ * NS_;
  float* kbuf = ws + o; o += (size_t)AROWS_ * R_;
  float* vbuf = ws + o; o += (size_t)AROWS_ * R_;
  float* rbuf = ws + o; o += (size_t)AROWS_ * R_;
  float* tc   = ws + o; o += (size_t)AROWS_ * C_;

  // ---------- f16 workspace ----------
  _Float16* hs = (_Float16*)(ws + o);
  size_t ho = 0;
  _Float16* xs_h  = hs + ho; ho += (size_t)AROWS_ * C_;
  _Float16* xd_t  = hs + ho; ho += (size_t)TROWS_ * R_;
  _Float16* xd_c  = hs + ho; ho += (size_t)CROWS_ * R_;
  _Float16* ybuf  = hs + ho; ho += (size_t)AROWS_ * R_;
  _Float16* ycat  = hs + ho; ho += (size_t)AROWS_ * R_;
  _Float16* qn    = hs + ho; ho += (size_t)AROWS_ * C_;
  _Float16* fn    = hs + ho; ho += (size_t)AROWS_ * C_;
  _Float16* qh    = hs + ho; ho += (size_t)AROWS_ * C_;
  _Float16* kh    = hs + ho; ho += (size_t)AROWS_ * C_;
  _Float16* vh    = hs + ho; ho += (size_t)AROWS_ * C_;
  _Float16* ob    = hs + ho; ho += (size_t)AROWS_ * C_;
  _Float16* wdo_h = hs + ho; ho += (size_t)R_ * C_;
  _Float16* wdc_h = hs + ho; ho += (size_t)R_ * C_;
  _Float16* wup_h = hs + ho; ho += (size_t)C_ * R_;
  _Float16* tek_h = hs + ho; ho += (size_t)R_ * R_;
  _Float16* tev_h = hs + ho; ho += (size_t)R_ * R_;
  _Float16* ter_h = hs + ho; ho += (size_t)R_ * R_;
  _Float16* teo_h = hs + ho; ho += (size_t)R_ * R_;
  _Float16* cek_h = hs + ho; ho += (size_t)R_ * R_;
  _Float16* cev_h = hs + ho; ho += (size_t)R_ * R_;
  _Float16* cer_h = hs + ho; ho += (size_t)R_ * R_;
  _Float16* ceo_h = hs + ho; ho += (size_t)R_ * R_;
  _Float16* win_h = hs + ho; ho += (size_t)3 * C_ * C_;
  _Float16* wout_h= hs + ho; ho += (size_t)C_ * C_;
  int* rm_t  = (int*)(hs + ho);
  int* rm_c  = rm_t + TROWS_;
  int* rmy_t = rm_c + CROWS_;
  int* rmy_c = rmy_t + TROWS_;

  float* out    = (float*)d_out;
  float* tp_out = out;
  float* xs_out = out + (size_t)B_ * 12 * C_;
  float* nst    = xs_out + (size_t)B_ * NS_ * C_;
  float* nsc    = nst + (size_t)3 * TROWS_ * R_;

  // 0) one-time f16 conversions (weights + xs)
  cvt_f16_kernel<<<cvt_blocks(AROWS_ * C_), 256, 0, stream>>>(xs, xs_h, AROWS_ * C_);
  cvt_f16_kernel<<<cvt_blocks(R_ * C_), 256, 0, stream>>>(Wdo_w, wdo_h, R_ * C_);
  cvt_f16_kernel<<<cvt_blocks(R_ * C_), 256, 0, stream>>>(Wdc_w, wdc_h, R_ * C_);
  cvt_f16_kernel<<<cvt_blocks(C_ * R_), 256, 0, stream>>>(Wup_w, wup_h, C_ * R_);
  cvt_f16_kernel<<<cvt_blocks(R_ * R_), 256, 0, stream>>>(te_k, tek_h, R_ * R_);
  cvt_f16_kernel<<<cvt_blocks(R_ * R_), 256, 0, stream>>>(te_v, tev_h, R_ * R_);
  cvt_f16_kernel<<<cvt_blocks(R_ * R_), 256, 0, stream>>>(te_r, ter_h, R_ * R_);
  cvt_f16_kernel<<<cvt_blocks(R_ * R_), 256, 0, stream>>>(te_o, teo_h, R_ * R_);
  cvt_f16_kernel<<<cvt_blocks(R_ * R_), 256, 0, stream>>>(ce_k, cek_h, R_ * R_);
  cvt_f16_kernel<<<cvt_blocks(R_ * R_), 256, 0, stream>>>(ce_v, cev_h, R_ * R_);
  cvt_f16_kernel<<<cvt_blocks(R_ * R_), 256, 0, stream>>>(ce_r, cer_h, R_ * R_);
  cvt_f16_kernel<<<cvt_blocks(R_ * R_), 256, 0, stream>>>(ce_o, ceo_h, R_ * R_);
  cvt_f16_kernel<<<cvt_blocks(3 * C_ * C_), 256, 0, stream>>>(attn_in_w, win_h, 3 * C_ * C_);
  cvt_f16_kernel<<<cvt_blocks(C_ * C_), 256, 0, stream>>>(attn_out_w, wout_h, C_ * C_);

  // 1) routing statistics
  colmean_kernel<<<dim3(B_, C_ / 128), 128, 0, stream>>>(xz, zn);
  l2n_kernel<<<B_, 256, 0, stream>>>(zn);
  resp_kernel<<<(B_ * NS_) / 8, 256, 0, stream>>>(xs, zn, resp);
  route_kernel<<<B_, 256, 0, stream>>>(resp, rm_t, rm_c, rmy_t, rmy_c);

  // 2) gathered down projections (f16 out)
  gemm_xwT_kernel<true, false, false, true><<<gemm_blocks(TROWS_, R_), 128, 0, stream>>>(
      xs_h, C_, rm_t, wdo_h, C_, Wdo_b, xd_t, R_, nullptr, TROWS_, R_, C_, nullptr, nullptr);
  gemm_xwT_kernel<true, false, false, true><<<gemm_blocks(CROWS_, R_), 128, 0, stream>>>(
      xs_h, C_, rm_c, wdc_h, C_, Wdc_b, xd_c, R_, nullptr, CROWS_, R_, C_, nullptr, nullptr);

  // 3) RWKV k/v/r projections (f32 out, feeds exp math)
  const int TOFF = 0, COFF = TROWS_ * R_;
  gemm_xwT_kernel<false, false, false, false><<<gemm_blocks(TROWS_, R_), 128, 0, stream>>>(
      xd_t, R_, nullptr, tek_h, R_, nullptr, kbuf + TOFF, R_, nullptr, TROWS_, R_, R_, nullptr, nullptr);
  gemm_xwT_kernel<false, false, false, false><<<gemm_blocks(TROWS_, R_), 128, 0, stream>>>(
      xd_t, R_, nullptr, tev_h, R_, nullptr, vbuf + TOFF, R_, nullptr, TROWS_, R_, R_, nullptr, nullptr);
  gemm_xwT_kernel<false, false, false, false><<<gemm_blocks(TROWS_, R_), 128, 0, stream>>>(
      xd_t, R_, nullptr, ter_h, R_, nullptr, rbuf + TOFF, R_, nullptr, TROWS_, R_, R_, nullptr, nullptr);
  gemm_xwT_kernel<false, false, false, false><<<gemm_blocks(CROWS_, R_), 128, 0, stream>>>(
      xd_c, R_, nullptr, cek_h, R_, nullptr, kbuf + COFF, R_, nullptr, CROWS_, R_, R_, nullptr, nullptr);
  gemm_xwT_kernel<false, false, false, false><<<gemm_blocks(CROWS_, R_), 128, 0, stream>>>(
      xd_c, R_, nullptr, cev_h, R_, nullptr, vbuf + COFF, R_, nullptr, CROWS_, R_, R_, nullptr, nullptr);
  gemm_xwT_kernel<false, false, false, false><<<gemm_blocks(CROWS_, R_), 128, 0, stream>>>(
      xd_c, R_, nullptr, cer_h, R_, nullptr, rbuf + COFF, R_, nullptr, CROWS_, R_, R_, nullptr, nullptr);

  // 4) RWKV elementwise (states straight into d_out; y as f16)
  rwkv_ew_kernel<<<(TROWS_ * R_ + 255) / 256, 256, 0, stream>>>(
      kbuf + TOFF, vbuf + TOFF, rbuf + TOFF, state_t, te_td, te_tf, ybuf + TOFF, nst, TROWS_);
  rwkv_ew_kernel<<<(CROWS_ * R_ + 255) / 256, 256, 0, stream>>>(
      kbuf + COFF, vbuf + COFF, rbuf + COFF, state_c, ce_td, ce_tf, ybuf + COFF, nsc, CROWS_);

  // 5) RWKV output projections, scattered into concatenated order (f16 out)
  gemm_xwT_kernel<false, true, false, true><<<gemm_blocks(TROWS_, R_), 128, 0, stream>>>(
      ybuf + TOFF, R_, nullptr, teo_h, R_, nullptr, ycat, R_, rmy_t, TROWS_, R_, R_, nullptr, nullptr);
  gemm_xwT_kernel<false, true, false, true><<<gemm_blocks(CROWS_, R_), 128, 0, stream>>>(
      ybuf + COFF, R_, nullptr, ceo_h, R_, nullptr, ycat, R_, rmy_c, CROWS_, R_, R_, nullptr, nullptr);

  // 6) up-projection (f32 out: feeds LN + t_p) and t_p slice
  gemm_xwT_kernel<false, false, false, false><<<gemm_blocks(AROWS_, C_), 128, 0, stream>>>(
      ycat, R_, nullptr, wup_h, R_, Wup_b, tc, C_, nullptr, AROWS_, C_, R_, nullptr, nullptr);
  copy_tp_kernel<<<(B_ * 12 * C_ + 255) / 256, 256, 0, stream>>>(tc, tp_out);

  // 7) LayerNorms (f16 out)
  ln_kernel<<<(AROWS_ + 7) / 8, 256, 0, stream>>>(xs, qn_g, qn_b, qn, AROWS_);
  ln_kernel<<<(AROWS_ + 7) / 8, 256, 0, stream>>>(tc, fn_g, fn_b, fn, AROWS_);

  // 8) QKV projections (f16 out -> attention)
  gemm_xwT_kernel<false, false, false, true><<<gemm_blocks(AROWS_, C_), 128, 0, stream>>>(
      qn, C_, nullptr, win_h, C_, attn_in_b, qh, C_, nullptr, AROWS_, C_, C_, nullptr, nullptr);
  gemm_xwT_kernel<false, false, false, true><<<gemm_blocks(AROWS_, C_), 128, 0, stream>>>(
      fn, C_, nullptr, win_h + (size_t)C_ * C_, C_, attn_in_b + C_, kh, C_, nullptr,
      AROWS_, C_, C_, nullptr, nullptr);
  gemm_xwT_kernel<false, false, false, true><<<gemm_blocks(AROWS_, C_), 128, 0, stream>>>(
      fn, C_, nullptr, win_h + (size_t)2 * C_ * C_, C_, attn_in_b + 2 * C_, vh, C_, nullptr,
      AROWS_, C_, C_, nullptr, nullptr);

  // 9) flash attention (f16 in/out)
  attn_kernel<<<B_ * NH_ * (NS_ / 16 / 4), 128, 0, stream>>>(qh, kh, vh, ob);

  // 10) output projection with residual: xs + router * (o @ W^T + b)  (f32 out)
  gemm_xwT_kernel<false, false, true, false><<<gemm_blocks(AROWS_, C_), 128, 0, stream>>>(
      ob, C_, nullptr, wout_h, C_, attn_out_b, xs_out, C_, nullptr,
      AROWS_, C_, C_, xs, router);
}
